// MultiDownAdapterMoE_26680336842992
// MI455X (gfx1250) — compile-verified, matrix operand order checked by reading, not run on hardware
//
#include <hip/hip_runtime.h>
#include <hip/hip_bf16.h>

// ---------------------------------------------------------------------------
// MultiDownAdapterMoE fused kernel for gfx1250 (MI455X), wave32 + WMMA bf16.
//
// Roofline: ~200MB traffic (x in + out) @ 23.3 TB/s ~= 8.6us; dense compute
// 24 GFLOP hidden under v_wmma_f32_16x16x32_bf16. Experts concatenated along
// the hidden dim (240 -> pad 256); per-token mask applied by zeroing H
// columns, making the up-projection one uniform GEMM.
// x tile staged to LDS via GLOBAL_LOAD_ASYNC_TO_LDS_B128 (ASYNCcnt path).
// ---------------------------------------------------------------------------

typedef __bf16  bf16x16 __attribute__((ext_vector_type(16)));
typedef __bf16  bf16x8  __attribute__((ext_vector_type(8)));
typedef float   f32x8   __attribute__((ext_vector_type(8)));

union V16 { bf16x16 v; bf16x8 h[2]; };

#define TOKENS      32768       // B*S = 16*2048
#define DMODEL      768
#define NPAD        256         // 240 expert cols padded to 256
#define MT          64          // tokens per workgroup
#define HSTR        264         // padded LDS row stride for H (bf16)

__device__ __forceinline__ f32x8 wmma_bf16(bf16x16 a, bf16x16 b, f32x8 c) {
    // (neg_a, A, neg_b, B, c_mod, C, reuse_a, reuse_b)
    return __builtin_amdgcn_wmma_f32_16x16x32_bf16(false, a, false, b,
                                                   (short)0, c, false, false);
}

// ---------------------------------------------------------------------------
// Pack kernel: build bf16, N-major (transposed) weight panels in workspace.
//   WdT [256][768]  : row n = concat expert column n (bf16), rows 240..255 = 0
//   WuT [768][256]  : row n(out col) over K=256 (expert-block rows), K>=240 = 0
//   bdp [256]       : concatenated down biases (f32), 240..255 = 0
// expert col blocks: e0:[0,16) e1:[16,48) e2:[48,112) e3:[112,240)
// ---------------------------------------------------------------------------
__global__ void pack_weights(
    const float* __restrict__ Wd0, const float* __restrict__ Wd1,
    const float* __restrict__ Wd2, const float* __restrict__ Wd3,
    const float* __restrict__ bd0, const float* __restrict__ bd1,
    const float* __restrict__ bd2, const float* __restrict__ bd3,
    const float* __restrict__ Wu0, const float* __restrict__ Wu1,
    const float* __restrict__ Wu2, const float* __restrict__ Wu3,
    __bf16* __restrict__ WdT, __bf16* __restrict__ WuT,
    float* __restrict__ bdp)
{
    int t = blockIdx.x * blockDim.x + threadIdx.x;
    if (t >= NPAD * DMODEL) return;

    // WdT element: n-major rows over K=768
    {
        int n = t / DMODEL, k = t % DMODEL;
        float v = 0.f;
        if      (n <  16) v = Wd0[k * 16  +  n       ];
        else if (n <  48) v = Wd1[k * 32  + (n -  16)];
        else if (n < 112) v = Wd2[k * 64  + (n -  48)];
        else if (n < 240) v = Wd3[k * 128 + (n - 112)];
        WdT[t] = (__bf16)v;
    }
    // WuT element: rows = output col (768), cols = K (256)
    {
        int n = t / NPAD, kk = t % NPAD;
        float v = 0.f;
        if      (kk <  16) v = Wu0[ kk        * DMODEL + n];
        else if (kk <  48) v = Wu1[(kk -  16) * DMODEL + n];
        else if (kk < 112) v = Wu2[(kk -  48) * DMODEL + n];
        else if (kk < 240) v = Wu3[(kk - 112) * DMODEL + n];
        WuT[t] = (__bf16)v;
    }
    if (t < NPAD) {
        float b = 0.f;
        if      (t <  16) b = bd0[t];
        else if (t <  48) b = bd1[t -  16];
        else if (t < 112) b = bd2[t -  48];
        else if (t < 240) b = bd3[t - 112];
        bdp[t] = b;
    }
}

// ---------------------------------------------------------------------------
// Fused MoE adapter kernel. 256 threads = 8 wave32 per workgroup, MT=64 tokens.
// Dynamic LDS: sX f32[64][768] | sH bf16[64][HSTR] | smask i32[64]
// ---------------------------------------------------------------------------
__global__ __launch_bounds__(256, 1)
void moe_fused(const float* __restrict__ x, const int* __restrict__ mask,
               const __bf16* __restrict__ WdT, const __bf16* __restrict__ WuT,
               const float* __restrict__ bdp,
               const float* __restrict__ bu0, const float* __restrict__ bu1,
               const float* __restrict__ bu2, const float* __restrict__ bu3,
               float* __restrict__ out)
{
    extern __shared__ char smem[];
    float*  sX    = (float*)smem;                                  // 196608 B
    __bf16* sH    = (__bf16*)(smem + MT * DMODEL * 4);             //  33792 B
    int*    smask = (int*)(smem + MT * DMODEL * 4 + MT * HSTR * 2);

    const int tid  = threadIdx.x;
    const int wave = tid >> 5;
    const int lane = tid & 31;
    const int lr   = lane & 15;      // col/row index within 16
    const int lh   = lane >> 4;      // half-wave selects K sub-pattern
    const int mw   = wave >> 1;      // M tile 0..3
    const int nh   = wave & 1;       // N half
    const int rb   = mw * 16;        // row base within tile
    const int t0   = blockIdx.x * MT;

    // ---- stage x tile (f32) into LDS via async direct-to-LDS copies ---------
    {
        const unsigned long long gbase =
            (unsigned long long)(x + (size_t)t0 * DMODEL);
        const unsigned lbase = (unsigned)(unsigned long long)sX;
        // 64*768 f32 = 196608 B; 256 threads x 16 B -> 48 b128 copies each
        for (int i = tid; i < MT * DMODEL / 4; i += 256) {
            unsigned long long ga = gbase + (unsigned long long)i * 16u;
            unsigned           la = lbase + (unsigned)i * 16u;
            asm volatile("global_load_async_to_lds_b128 %0, %1, off"
                         :: "v"(la), "v"(ga) : "memory");
        }
        // warm L2 with the GEMM2 weight panel while the async fill runs
        __builtin_prefetch(WuT + (size_t)tid * 768, 0, 1);  // 256 x 1536B spans
        asm volatile("s_wait_asynccnt 0" ::: "memory");
    }
    if (tid < MT) smask[tid] = mask[t0 + tid];
    __syncthreads();

    // experts for the 8 rows this lane's accumulator elements touch
    int ei[8];
    #pragma unroll
    for (int i = 0; i < 8; ++i) ei[i] = smask[rb + 8 * lh + i];

    // ---- GEMM1: H[64,256] = relu(X·WdT + bd), masked to selected block -------
    f32x8 acc[8];
    #pragma unroll
    for (int j = 0; j < 8; ++j)
        acc[j] = (f32x8){0.f, 0.f, 0.f, 0.f, 0.f, 0.f, 0.f, 0.f};

    for (int k0 = 0; k0 < DMODEL; k0 += 32) {
        // A fragment: 16x32 bf16 from f32 LDS (ISA K-striping per half-wave)
        const float* ap = sX + (rb + lr) * DMODEL + k0 + lh * 8;
        V16 a;
        #pragma unroll
        for (int i = 0; i < 8; ++i) a.v[i]     = (__bf16)ap[i];
        #pragma unroll
        for (int i = 0; i < 8; ++i) a.v[8 + i] = (__bf16)ap[16 + i];

        #pragma unroll
        for (int j = 0; j < 8; ++j) {
            int n = (nh * 8 + j) * 16 + lr;     // WdT row (expert col)
            const bf16x8* bp =
                (const bf16x8*)(WdT + (size_t)n * DMODEL + k0 + lh * 8);
            V16 b; b.h[0] = bp[0]; b.h[1] = bp[2];
            acc[j] = wmma_bf16(a.v, b.v, acc[j]);
        }
    }

    // epilogue 1: bias + relu + branchless expert-block mask -> sH (bf16)
    #pragma unroll
    for (int j = 0; j < 8; ++j) {
        int n = (nh * 8 + j) * 16 + lr;
        float bdv = bdp[n];
        #pragma unroll
        for (int i = 0; i < 8; ++i) {
            int e        = ei[i];
            int offE     = (16 << e) - 16;      // 0,16,48,112
            unsigned dsE = (unsigned)(16 << e); // 16,32,64,128
            float v = fmaxf(acc[j][i] + bdv, 0.f);
            v = ((unsigned)(n - offE) < dsE) ? v : 0.f;   // single cmp+cndmask
            sH[(rb + 8 * lh + i) * HSTR + n] = (__bf16)v;
        }
    }
    __syncthreads();

    // ---- GEMM2: out[64,768] = H·WuT + bu[e] + x ------------------------------
    for (int nc = 0; nc < 6; ++nc) {            // 6 chunks x 128 cols
        f32x8 acc2[4];
        #pragma unroll
        for (int j = 0; j < 4; ++j)
            acc2[j] = (f32x8){0.f, 0.f, 0.f, 0.f, 0.f, 0.f, 0.f, 0.f};

        #pragma unroll
        for (int kk = 0; kk < 8; ++kk) {        // K = 256
            int k0 = kk * 32;
            const __bf16* hp = sH + (rb + lr) * HSTR + k0 + lh * 8;
            V16 a;
            a.h[0] = *(const bf16x8*)hp;
            a.h[1] = *(const bf16x8*)(hp + 16);

            #pragma unroll
            for (int j = 0; j < 4; ++j) {
                int n = (nc * 8 + nh * 4 + j) * 16 + lr;   // output col
                const bf16x8* bp =
                    (const bf16x8*)(WuT + (size_t)n * NPAD + k0 + lh * 8);
                V16 b; b.h[0] = bp[0]; b.h[1] = bp[2];
                acc2[j] = wmma_bf16(a.v, b.v, acc2[j]);
            }
        }

        // epilogue 2: + bu[e(token)] + residual, store f32
        #pragma unroll
        for (int j = 0; j < 4; ++j) {
            int n = (nc * 8 + nh * 4 + j) * 16 + lr;
            float b0 = bu0[n], b1 = bu1[n], b2 = bu2[n], b3 = bu3[n];
            #pragma unroll
            for (int i = 0; i < 8; ++i) {
                int row = rb + 8 * lh + i;
                int e   = ei[i];
                float buv = (e == 0) ? b0 : (e == 1) ? b1 : (e == 2) ? b2 : b3;
                float res = sX[row * DMODEL + n];
                out[(size_t)(t0 + row) * DMODEL + n] = acc2[j][i] + buv + res;
            }
        }
    }
}

// ---------------------------------------------------------------------------
extern "C" void kernel_launch(void* const* d_in, const int* in_sizes, int n_in,
                              void* d_out, int out_size, void* d_ws, size_t ws_size,
                              hipStream_t stream)
{
    (void)in_sizes; (void)n_in; (void)out_size; (void)ws_size;

    const float* x    = (const float*)d_in[0];
    const int*   mask = (const int*)d_in[1];
    const float* Wd0  = (const float*)d_in[2];
    const float* bd0  = (const float*)d_in[3];
    const float* Wu0  = (const float*)d_in[4];
    const float* bu0  = (const float*)d_in[5];
    const float* Wd1  = (const float*)d_in[6];
    const float* bd1  = (const float*)d_in[7];
    const float* Wu1  = (const float*)d_in[8];
    const float* bu1  = (const float*)d_in[9];
    const float* Wd2  = (const float*)d_in[10];
    const float* bd2  = (const float*)d_in[11];
    const float* Wu2  = (const float*)d_in[12];
    const float* bu2  = (const float*)d_in[13];
    const float* Wd3  = (const float*)d_in[14];
    const float* bd3  = (const float*)d_in[15];
    const float* Wu3  = (const float*)d_in[16];
    const float* bu3  = (const float*)d_in[17];

    __bf16* WdT = (__bf16*)d_ws;                 // [256][768] bf16
    __bf16* WuT = WdT + NPAD * DMODEL;           // [768][256] bf16
    float*  bdp = (float*)(WuT + DMODEL * NPAD); // [256] f32

    // 1) pack/transpose/convert weights (cheap: ~400K elements)
    pack_weights<<<(NPAD * DMODEL + 255) / 256, 256, 0, stream>>>(
        Wd0, Wd1, Wd2, Wd3, bd0, bd1, bd2, bd3,
        Wu0, Wu1, Wu2, Wu3, WdT, WuT, bdp);

    // 2) fused MoE adapter
    size_t lds_bytes = MT * DMODEL * 4 + MT * HSTR * 2 + MT * 4; // ~230.7 KB
    moe_fused<<<TOKENS / MT, 256, lds_bytes, stream>>>(
        x, mask, WdT, WuT, bdp, bu0, bu1, bu2, bu3, (float*)d_out);
}
